// AttentionHead_8315056685234
// MI455X (gfx1250) — compile-verified, matrix operand order checked by reading, not run on hardware
//
#include <hip/hip_runtime.h>

typedef __attribute__((ext_vector_type(16))) __bf16 v16bf;
typedef __attribute__((ext_vector_type(8)))  float  v8f;

#define EMB 128
#define LDP 136   // padded LDS row stride (elements): 272B = 17*16B -> conflict-free b128
#define VLP 72    // padded stride for V^T / P tiles: 144B = 9*16B

union Frag { v16bf v; uint4 q[2]; };

__device__ __forceinline__ unsigned short f2bf(float f) {
  union { float f; unsigned u; } x; x.f = f;
  unsigned u = x.u + 0x7FFFu + ((x.u >> 16) & 1u);  // round-to-nearest-even
  return (unsigned short)(u >> 16);
}
__device__ __forceinline__ unsigned pack2(float a, float b) {
  return (unsigned)f2bf(a) | ((unsigned)f2bf(b) << 16);
}
// Load a 16x32 bf16 A/B fragment slice for this lane: 8 elems at p, 8 at p+16.
__device__ __forceinline__ v16bf ldfrag(const unsigned short* p) {
  Frag f;
  f.q[0] = *(const uint4*)(p);
  f.q[1] = *(const uint4*)(p + 16);
  return f.v;
}

// CDNA5 async global->LDS copy, 16B per lane, tracked with ASYNCcnt.
// LDS dest offset = low 32 bits of the generic pointer into __shared__.
__device__ __forceinline__ void async_copy16(void* lds_dst, const void* gsrc) {
  unsigned off = (unsigned)(unsigned long long)lds_dst;
  asm volatile("global_load_async_to_lds_b128 %0, %1, off"
               :: "v"(off), "v"(gsrc) : "memory");
}
__device__ __forceinline__ void wait_async0() {
  asm volatile("s_wait_asynccnt 0" ::: "memory");
}

// ---------------------------------------------------------------------------
// Kernel 1: Q/K/V = X @ W^T + b, output bf16. V written transposed per batch.
// Grid: 256 blocks x 256 threads (8 waves). Each block: 64 rows of X.
// Wave (rgrp,cgrp): 16 rows x 64 cols of the 64x128 output tile.
// ---------------------------------------------------------------------------
__global__ __launch_bounds__(256) void qkv_kernel(
    const float* __restrict__ X,
    const float* __restrict__ Wq, const float* __restrict__ bq,
    const float* __restrict__ Wk, const float* __restrict__ bk,
    const float* __restrict__ Wv, const float* __restrict__ bv,
    unsigned short* __restrict__ Qb, unsigned short* __restrict__ Kb,
    unsigned short* __restrict__ Vt)
{
  __shared__ __align__(16) unsigned short Xl[64][LDP];
  __shared__ __align__(16) unsigned short Wl[128][LDP];

  const int tid  = threadIdx.x;
  const int lane = tid & 31;
  const int wave = tid >> 5;
  const int half = (lane >> 4) & 1;
  const int l16  = lane & 15;
  const int rgrp = wave >> 1;           // 0..3: 16-row group
  const int cgrp = wave & 1;            // 0..1: 64-col group
  const long rowBase = (long)blockIdx.x * 64;

  // Stage X tile (64x128 f32 -> bf16 LDS); conversion requires VGPR round trip
  for (int i = tid; i < 64 * EMB / 4; i += 256) {
    float4 x = ((const float4*)(X + rowBase * EMB))[i];
    int r = (i * 4) >> 7, c = (i * 4) & 127;
    *(unsigned*)&Xl[r][c]     = pack2(x.x, x.y);
    *(unsigned*)&Xl[r][c + 2] = pack2(x.z, x.w);
  }

  for (int m = 0; m < 3; ++m) {
    const float* W    = (m == 0) ? Wq : (m == 1) ? Wk : Wv;
    const float* bias = (m == 0) ? bq : (m == 1) ? bk : bv;
    __syncthreads();
    for (int i = tid; i < EMB * EMB / 4; i += 256) {
      float4 w = ((const float4*)W)[i];
      int r = (i * 4) >> 7, c = (i * 4) & 127;
      *(unsigned*)&Wl[r][c]     = pack2(w.x, w.y);
      *(unsigned*)&Wl[r][c + 2] = pack2(w.z, w.w);
    }
    __syncthreads();

    v8f acc[4] = {};
#pragma unroll
    for (int k0 = 0; k0 < EMB; k0 += 32) {
      v16bf a = ldfrag(&Xl[rgrp * 16 + l16][k0 + half * 8]);
#pragma unroll
      for (int t = 0; t < 4; ++t) {
        v16bf bm = ldfrag(&Wl[(cgrp * 4 + t) * 16 + l16][k0 + half * 8]);
        acc[t] = __builtin_amdgcn_wmma_f32_16x16x32_bf16(
            false, a, false, bm, (short)0, acc[t], false, false);
      }
    }

#pragma unroll
    for (int t = 0; t < 4; ++t) {
      int col = (cgrp * 4 + t) * 16 + l16;
      float bval = bias[col];
#pragma unroll
      for (int j = 0; j < 8; ++j) {
        long grow = rowBase + rgrp * 16 + j + half * 8;  // global row in [0,16384)
        unsigned short h = f2bf(acc[t][j] + bval);
        if (m == 0)      Qb[grow * EMB + col] = h;
        else if (m == 1) Kb[grow * EMB + col] = h;
        else {
          long batch = grow >> 11, s = grow & 2047;
          Vt[(batch * EMB + col) * 2048 + s] = h;       // V transposed: [b][d][S]
        }
      }
    }
  }
}

// ---------------------------------------------------------------------------
// Kernel 2: flash attention. Grid (32 q-tiles, 8 batches) x 128 thr (4 waves).
// Each wave: 16 query rows, full D=128; loop over 32 key chunks of 64.
// Staging uses CDNA5 async global->LDS (ASYNCcnt) to keep the VALU free.
// ---------------------------------------------------------------------------
__global__ __launch_bounds__(128) void attn_kernel(
    const unsigned short* __restrict__ Qb,
    const unsigned short* __restrict__ Kb,
    const unsigned short* __restrict__ Vt,
    float* __restrict__ Out)
{
  __shared__ __align__(16) unsigned short Ql[64][LDP];
  __shared__ __align__(16) unsigned short Kl[64][LDP];
  __shared__ __align__(16) unsigned short Vl[128][VLP];   // [d][key]
  __shared__ __align__(16) unsigned short Pl[4][16][VLP]; // wave-private P scratch

  const int tid = threadIdx.x, lane = tid & 31, wave = tid >> 5;
  const int half = (lane >> 4) & 1, l16 = lane & 15;
  const int b = blockIdx.y;
  const long qrow0 = (long)b * 2048 + (long)blockIdx.x * 64;

  // Stage Q tile once (64x128 bf16), async: 8 instructions/wave, 512B each
  for (int i = tid; i < 64 * EMB / 8; i += 128) {
    int r = i >> 4, c = (i & 15) * 8;
    async_copy16(&Ql[r][c], Qb + qrow0 * EMB + (long)i * 8);
  }

  float mrow[8], lrow[8], alpha[8];
  v8f acco[8] = {};
#pragma unroll
  for (int j = 0; j < 8; ++j) { mrow[j] = -3.0e38f; lrow[j] = 0.0f; }

  const float scale = 0.088388347648318447f;  // 1/sqrt(128)

  for (int kc = 0; kc < 32; ++kc) {
    const long kbase = (long)kc * 64;
    __syncthreads();  // previous chunk's LDS reads complete in all waves
    // Stage K chunk (64 keys x 128) row-major, async
    for (int i = tid; i < 64 * EMB / 8; i += 128) {
      int r = i >> 4, c = (i & 15) * 8;
      async_copy16(&Kl[r][c],
                   Kb + ((long)b * 2048 + kbase) * EMB + (long)i * 8);
    }
    // Stage V^T chunk (128 d x 64 keys), async
    for (int i = tid; i < EMB * 64 / 8; i += 128) {
      int d = i >> 3, c = (i & 7) * 8;
      async_copy16(&Vl[d][c],
                   Vt + ((long)b * EMB + d) * 2048 + kbase + c);
    }
    if (kc + 1 < 32)
      __builtin_prefetch(Kb + ((long)b * 2048 + kbase + 64) * EMB, 0, 0);
    wait_async0();    // all async LDS writes (incl. Q on iter 0) complete
    __syncthreads();

    // S = Q K^T : 4 key n-tiles x 4 K-dim iters = 16 WMMA
    v8f accs[4] = {};
#pragma unroll
    for (int k0 = 0; k0 < EMB; k0 += 32) {
      v16bf a = ldfrag(&Ql[wave * 16 + l16][k0 + half * 8]);
#pragma unroll
      for (int t = 0; t < 4; ++t) {
        v16bf bm = ldfrag(&Kl[t * 16 + l16][k0 + half * 8]);
        accs[t] = __builtin_amdgcn_wmma_f32_16x16x32_bf16(
            false, a, false, bm, (short)0, accs[t], false, false);
      }
    }

    // Online softmax: row = VGPR slot j + half*8; columns across 16-lane half
#pragma unroll
    for (int j = 0; j < 8; ++j) {
      float mx = accs[0][j] * scale;
#pragma unroll
      for (int t = 1; t < 4; ++t) mx = fmaxf(mx, accs[t][j] * scale);
      mx = fmaxf(mx, __shfl_xor(mx, 1, 32));
      mx = fmaxf(mx, __shfl_xor(mx, 2, 32));
      mx = fmaxf(mx, __shfl_xor(mx, 4, 32));
      mx = fmaxf(mx, __shfl_xor(mx, 8, 32));
      float mn = fmaxf(mrow[j], mx);
      alpha[j] = __expf(mrow[j] - mn);
      mrow[j] = mn;
      float s = 0.0f;
#pragma unroll
      for (int t = 0; t < 4; ++t) {
        float p = __expf(accs[t][j] * scale - mn);
        accs[t][j] = p;
        s += p;
      }
      s += __shfl_xor(s, 1, 32);
      s += __shfl_xor(s, 2, 32);
      s += __shfl_xor(s, 4, 32);
      s += __shfl_xor(s, 8, 32);
      lrow[j] = lrow[j] * alpha[j] + s;
    }

    // P (C-layout) -> bf16 LDS (row-major) for re-read as WMMA A operand
#pragma unroll
    for (int t = 0; t < 4; ++t)
#pragma unroll
      for (int j = 0; j < 8; ++j)
        Pl[wave][j + half * 8][t * 16 + l16] = f2bf(accs[t][j]);

    // Rescale running O
#pragma unroll
    for (int n = 0; n < 8; ++n)
#pragma unroll
      for (int j = 0; j < 8; ++j)
        acco[n][j] *= alpha[j];

    // O += P @ V : 2 K-dim iters x 8 d n-tiles = 16 WMMA
#pragma unroll
    for (int kk = 0; kk < 2; ++kk) {
      v16bf a = ldfrag(&Pl[wave][l16][kk * 32 + half * 8]);
#pragma unroll
      for (int n = 0; n < 8; ++n) {
        v16bf bm = ldfrag(&Vl[n * 16 + l16][kk * 32 + half * 8]);
        acco[n] = __builtin_amdgcn_wmma_f32_16x16x32_bf16(
            false, a, false, bm, (short)0, acco[n], false, false);
      }
    }
  }

  // Epilogue: O / l, fp32 out
#pragma unroll
  for (int n = 0; n < 8; ++n) {
    int col = n * 16 + l16;
#pragma unroll
    for (int j = 0; j < 8; ++j) {
      long row = qrow0 + wave * 16 + j + half * 8;
      Out[row * EMB + col] = acco[n][j] / lrow[j];
    }
  }
}

extern "C" void kernel_launch(void* const* d_in, const int* in_sizes, int n_in,
                              void* d_out, int out_size, void* d_ws, size_t ws_size,
                              hipStream_t stream) {
  (void)in_sizes; (void)n_in; (void)out_size; (void)ws_size;
  const float* X  = (const float*)d_in[0];
  const float* Wq = (const float*)d_in[1];
  const float* bq = (const float*)d_in[2];
  const float* Wk = (const float*)d_in[3];
  const float* bk = (const float*)d_in[4];
  const float* Wv = (const float*)d_in[5];
  const float* bv = (const float*)d_in[6];

  unsigned short* Qb = (unsigned short*)d_ws;                 // [16384][128] bf16
  unsigned short* Kb = Qb + (size_t)16384 * 128;              // [16384][128] bf16
  unsigned short* Vt = Kb + (size_t)16384 * 128;              // [8][128][2048] bf16
  float* Out = (float*)d_out;

  qkv_kernel<<<256, 256, 0, stream>>>(X, Wq, bq, Wk, bk, Wv, bv, Qb, Kb, Vt);
  attn_kernel<<<dim3(32, 8), 128, 0, stream>>>(Qb, Kb, Vt, Out);
}
